// Decoder_35682588295654
// MI455X (gfx1250) — compile-verified, matrix operand order checked by reading, not run on hardware
//
#include <hip/hip_runtime.h>

// ---------------- problem constants ----------------
#define BB   32
#define TH   48
#define TF   24
#define CC   512
#define FF   8
#define HID  128
#define EMB  16
#define G3   384          // 3*HID
#define CELLS 8           // cells (b,c pairs) per workgroup
#define NWG  (BB * (CC / CELLS))   // 2048
#define NTHR 256

// swizzled-weight layout offsets (in ushort elements) inside d_ws
#define WHH_SW_OFF 0            // 24 ntiles * 4 kb * 32 lanes * 16 = 49152
#define WS_SW_OFF  49152        // 8 * 4 * 32 * 16 = 16384
#define WH_SW_OFF  65536        // 8 * 4 * 32 * 16 = 16384
#define SW_TOTAL   81920

// dynamic LDS bytes (see carve-up below)
#define SMEM_BYTES 274496

typedef __attribute__((ext_vector_type(16))) __bf16       bh16;
typedef __attribute__((ext_vector_type(8)))  float        f32x8;
typedef __attribute__((ext_vector_type(4)))  unsigned int u32x4;

union BV { bh16 bf; u32x4 q[2]; };

__device__ __forceinline__ unsigned short f2bf(float f) {
    unsigned u = __builtin_bit_cast(unsigned, f);
    u += 0x7FFFu + ((u >> 16) & 1u);          // round-to-nearest-even
    return (unsigned short)(u >> 16);
}
__device__ __forceinline__ float bf2f(unsigned short h) {
    return __builtin_bit_cast(float, (unsigned)h << 16);
}
// 2-transcendental tanh: v_exp_f32 + v_rcp_f32; exact at both saturated ends.
__device__ __forceinline__ float tanh_fast(float x) {
    float e = __expf(2.0f * x);
    return 1.0f - 2.0f * __builtin_amdgcn_rcpf(e + 1.0f);
}
__device__ __forceinline__ float sigmoid_fast(float x) {
    return __builtin_amdgcn_rcpf(1.0f + __expf(-x));
}

// ---------------------------------------------------------------------------
// Prep kernel: convert W_hh / attn_Ws / attn_Wh to bf16, pre-swizzled into the
// wave32 WMMA B-operand layout: for block (ntile, kb), lane l holds 16
// contiguous bf16 = B[k][n] with n = ntile*16 + (l&15),
// k = kb*32 + (l>=16 ? 8 : 0) + (e<8 ? e : e+8).
// ---------------------------------------------------------------------------
__global__ void prep_swizzle(const float* __restrict__ Whh,
                             const float* __restrict__ Ws,
                             const float* __restrict__ Wh,
                             unsigned short* __restrict__ ws) {
    int gid = blockIdx.x * NTHR + threadIdx.x;
    if (gid >= SW_TOTAL) return;
    int mat, rem;
    if (gid < WS_SW_OFF)      { mat = 0; rem = gid; }             // W_hh (B[k][n] = Whh[n][k])
    else if (gid < WH_SW_OFF) { mat = 1; rem = gid - WS_SW_OFF; } // attn_Ws (B[k][n] = Ws[k][n])
    else                      { mat = 2; rem = gid - WH_SW_OFF; } // attn_Wh (B[k][n] = Wh[k][n])
    int blk  = rem >> 9;            // (ntile*4 + kb)
    int lane = (rem >> 4) & 31;
    int e    = rem & 15;
    int tn   = blk >> 2, kb = blk & 3;
    int n = tn * 16 + (lane & 15);
    int k = kb * 32 + ((lane & 16) ? 8 : 0) + (e < 8 ? e : e + 8);
    float v;
    if (mat == 0)      v = Whh[n * HID + k];
    else if (mat == 1) v = Ws[k * HID + n];
    else               v = Wh[k * HID + n];
    ws[gid] = f2bf(v);
}

// ---------------------------------------------------------------------------
// Main kernel: one workgroup = 8 cells (same b, 8 consecutive c).
// Keeps bf16 H-tile and HWh-tile resident in LDS; runs all 24 scan steps.
// HWhs is stored TRANSPOSED ([n][row]) so WMMA C-tiles store as ds_store_b128.
// ---------------------------------------------------------------------------
__global__ __launch_bounds__(NTHR, 1)
void decoder_step_kernel(const float* __restrict__ X,
                         const float* __restrict__ H,
                         const float* __restrict__ xn0,
                         const float* __restrict__ emb_table,
                         const float* __restrict__ Wih,
                         const float* __restrict__ bih,
                         const float* __restrict__ bhh,
                         const float* __restrict__ attn_v,
                         const float* __restrict__ fcw,
                         const float* __restrict__ fcb,
                         const unsigned short* __restrict__ wsw,
                         float* __restrict__ out) {
    extern __shared__ unsigned char smem[];
    // ---- LDS carve-up (274496 bytes) ----
    unsigned short* Hs    = (unsigned short*)smem;       // [8][48][128] bf16 (row-major)
    unsigned short* HWhs  = Hs   + CELLS * TH * HID;     // [128][384] bf16 (TRANSPOSED)
    unsigned short* hnb   = HWhs + CELLS * TH * HID;     // [16][128] bf16 (carry h, padded)
    unsigned short* hnewb = hnb  + 16 * HID;             // [16][128] bf16 (hn_new, padded)
    float* hnf   = (float*)(hnewb + 16 * HID);           // [8][128] carry h f32
    float* ghs   = hnf  + CELLS * HID;                   // [8][384]
    float* gis   = ghs  + CELLS * G3;                    // [8][384]
    float* qs    = gis  + CELLS * G3;                    // [8][128]
    float* scs   = qs   + CELLS * HID;                   // [8][48]
    float* alps  = scs  + CELLS * TH;                    // [8][48]
    float* ctxs  = alps + CELLS * TH;                    // [8][128]
    float* vls   = ctxs + CELLS * HID;                   // [128]
    float* bihs  = vls  + HID;                           // [384]
    float* bhhs  = bihs + G3;                            // [384]
    float* wihs  = bhhs + G3;                            // [384][17]
    float* xns   = wihs + G3 * (EMB + 1);                // [8]
    int*   eidx  = (int*)(xns + CELLS);                  // [8]

    const unsigned short* Whh_sw = wsw + WHH_SW_OFF;
    const unsigned short* Ws_sw  = wsw + WS_SW_OFF;
    const unsigned short* Wh_sw  = wsw + WH_SW_OFF;

    const int b    = blockIdx.x >> 6;          // CC/CELLS = 64 groups per batch
    const int c0   = (blockIdx.x & 63) << 3;
    const int tid  = threadIdx.x;
    const int lane = tid & 31;
    const int wave = tid >> 5;
    const int koff0 = (lane & 16) ? 8 : 0;     // wave32 A/B K-split per ISA layout

    // ---- hoist loop-invariant B fragments for gh (3 tiles) and q (1 tile) ----
    BV Bgh[3][4], Bq[4];
    #pragma unroll
    for (int ti = 0; ti < 3; ++ti) {
        int nt = wave + ti * 8;
        #pragma unroll
        for (int kb = 0; kb < 4; ++kb) {
            const unsigned short* bp = Whh_sw + (((nt * 4 + kb) * 32 + lane) << 4);
            Bgh[ti][kb].q[0] = *(const u32x4*)bp;
            Bgh[ti][kb].q[1] = *(const u32x4*)(bp + 8);
        }
    }
    #pragma unroll
    for (int kb = 0; kb < 4; ++kb) {
        const unsigned short* bp = Ws_sw + (((wave * 4 + kb) * 32 + lane) << 4);
        Bq[kb].q[0] = *(const u32x4*)bp;
        Bq[kb].q[1] = *(const u32x4*)(bp + 8);
    }

    // ---- Phase 0: small constants -> LDS ----
    for (int i = tid; i < HID; i += NTHR) vls[i] = attn_v[i];
    for (int i = tid; i < G3;  i += NTHR) { bihs[i] = bih[i]; bhhs[i] = bhh[i]; }
    for (int i = tid; i < G3 * (EMB + 1); i += NTHR) wihs[i] = Wih[i];

    // ---- Phase 1: stream H tile -> LDS bf16 (packed b32 stores) ----
    for (int i = tid; i < CELLS * TH * HID / 2; i += NTHR) {
        int cell = i / (TH * HID / 2);
        int rem  = i - cell * (TH * HID / 2);
        int t = rem >> 6, h = (rem & 63) * 2;
        const float* src = H + (((size_t)b * TH + t) * CC + c0 + cell) * HID + h;
        unsigned pk = (unsigned)f2bf(src[0]) | ((unsigned)f2bf(src[1]) << 16);
        *(unsigned*)(Hs + (cell * TH + t) * HID + h) = pk;
    }
    for (int i = tid; i < 16 * HID / 2; i += NTHR) {
        int r = i >> 6, h = (i & 63) * 2;
        float h0 = 0.0f, h1 = 0.0f;
        if (r < CELLS) {
            const float* src = H + (((size_t)b * TH + (TH - 1)) * CC + c0 + r) * HID + h;
            h0 = src[0]; h1 = src[1];
            hnf[r * HID + h]     = h0;
            hnf[r * HID + h + 1] = h1;
        }
        unsigned pk = (unsigned)f2bf(h0) | ((unsigned)f2bf(h1) << 16);
        *(unsigned*)(hnb   + r * HID + h) = pk;
        *(unsigned*)(hnewb + r * HID + h) = 0u;    // padded rows stay zero forever
    }
    if (tid < CELLS) xns[tid] = xn0[(size_t)b * CC + c0 + tid];
    __syncthreads();

    // ---- Phase 2: HWh = H @ attn_Wh (384x128x128, WMMA bf16, b128 C-stores) --
    {
        const int n0 = wave * 16;
        BV Bv[4];
        #pragma unroll
        for (int kb = 0; kb < 4; ++kb) {
            const unsigned short* bp = Wh_sw + (((wave * 4 + kb) * 32 + lane) << 4);
            Bv[kb].q[0] = *(const u32x4*)bp;
            Bv[kb].q[1] = *(const u32x4*)(bp + 8);
        }
        for (int mt = 0; mt < 24; ++mt) {
            const unsigned short* rp = Hs + (mt * 16 + (lane & 15)) * HID;
            f32x8 acc = {};
            #pragma unroll
            for (int kb = 0; kb < 4; ++kb) {
                BV Av;
                Av.q[0] = *(const u32x4*)(rp + kb * 32 + koff0);
                Av.q[1] = *(const u32x4*)(rp + kb * 32 + koff0 + 16);
                acc = __builtin_amdgcn_wmma_f32_16x16x32_bf16(
                        false, Av.bf, false, Bv[kb].bf, (short)0, acc, false, false);
            }
            // transposed store: lane owns column col, rows mbase..mbase+7 contiguous
            int mbase = mt * 16 + ((lane & 16) ? 8 : 0);
            int col   = n0 + (lane & 15);
            u32x4 pk;
            #pragma unroll
            for (int g2 = 0; g2 < 4; ++g2)
                pk[g2] = (unsigned)f2bf(acc[2 * g2]) | ((unsigned)f2bf(acc[2 * g2 + 1]) << 16);
            *(u32x4*)(HWhs + (size_t)col * (CELLS * TH) + mbase) = pk;
        }
    }
    __syncthreads();

    // ==================== scan over TF = 24 steps ====================
    for (int step = 0; step < TF; ++step) {
        // -- gather embedding indices for this step --
        if (tid < CELLS) {
            float fx = X[(((size_t)b * (TH + TF) + step) * CC + c0 + tid) * FF + (FF - 1)];
            eidx[tid] = (int)fx;
        }
        __syncthreads();

        // -- gi = [xn, emb] @ W_ih^T + b_ih  (tiny K=17 -> VALU) --
        for (int o = tid; o < CELLS * G3; o += NTHR) {
            int cell = o / G3, n = o - cell * G3;
            const float* wr = wihs + n * (EMB + 1);
            const float* er = emb_table + (size_t)eidx[cell] * EMB;
            float s = wr[0] * xns[cell];
            #pragma unroll
            for (int j = 0; j < EMB; ++j) s += wr[1 + j] * er[j];
            gis[cell * G3 + n] = s + bihs[n];
        }

        // -- gh = hn @ W_hh^T  (16x384x128 WMMA; 3 n-tiles per wave) --
        {
            const unsigned short* rp = hnb + (lane & 15) * HID;
            BV Av[4];
            #pragma unroll
            for (int kb = 0; kb < 4; ++kb) {
                Av[kb].q[0] = *(const u32x4*)(rp + kb * 32 + koff0);
                Av[kb].q[1] = *(const u32x4*)(rp + kb * 32 + koff0 + 16);
            }
            #pragma unroll
            for (int ti = 0; ti < 3; ++ti) {
                int nt = wave + ti * 8;
                f32x8 acc = {};
                #pragma unroll
                for (int kb = 0; kb < 4; ++kb)
                    acc = __builtin_amdgcn_wmma_f32_16x16x32_bf16(
                            false, Av[kb].bf, false, Bgh[ti][kb].bf, (short)0, acc, false, false);
                int mbase = (lane & 16) ? 8 : 0;
                int col   = nt * 16 + (lane & 15);
                #pragma unroll
                for (int g = 0; g < 8; ++g) {
                    int m = mbase + g;
                    if (m < CELLS) ghs[m * G3 + col] = acc[g];
                }
            }
        }
        __syncthreads();

        // -- GRU elementwise -> hn_new (bf16 pairs, packed b32 stores) --
        for (int e = tid; e < CELLS * HID / 2; e += NTHR) {
            int cell = e >> 6, h = (e & 63) * 2;
            unsigned pk = 0;
            #pragma unroll
            for (int j = 0; j < 2; ++j) {
                int hh = h + j;
                float ir  = gis[cell * G3 + hh];
                float iz  = gis[cell * G3 + HID + hh];
                float in_ = gis[cell * G3 + 2 * HID + hh];
                float hr  = ghs[cell * G3 + hh]           + bhhs[hh];
                float hz  = ghs[cell * G3 + HID + hh]     + bhhs[HID + hh];
                float hn_ = ghs[cell * G3 + 2 * HID + hh] + bhhs[2 * HID + hh];
                float r = sigmoid_fast(ir + hr);
                float z = sigmoid_fast(iz + hz);
                float nn = tanh_fast(in_ + r * hn_);
                float hv = (1.0f - z) * nn + z * hnf[cell * HID + hh];
                pk |= ((unsigned)f2bf(hv)) << (16 * j);
            }
            *(unsigned*)(hnewb + cell * HID + h) = pk;
        }
        __syncthreads();

        // -- q = hn_new @ attn_Ws  (16x128x128 WMMA; 1 n-tile per wave) --
        {
            const unsigned short* rp = hnewb + (lane & 15) * HID;
            f32x8 acc = {};
            #pragma unroll
            for (int kb = 0; kb < 4; ++kb) {
                BV Av;
                Av.q[0] = *(const u32x4*)(rp + kb * 32 + koff0);
                Av.q[1] = *(const u32x4*)(rp + kb * 32 + koff0 + 16);
                acc = __builtin_amdgcn_wmma_f32_16x16x32_bf16(
                        false, Av.bf, false, Bq[kb].bf, (short)0, acc, false, false);
            }
            int mbase = (lane & 16) ? 8 : 0;
            int col   = wave * 16 + (lane & 15);
            #pragma unroll
            for (int g = 0; g < 8; ++g) {
                int m = mbase + g;
                if (m < CELLS) qs[m * HID + col] = acc[g];
            }
        }
        __syncthreads();

        // -- attention scores: s[cell][t] = sum_h tanh(HWh + q) * v --
        for (int p = tid; p < CELLS * TH; p += NTHR) {
            int cell = p / TH, t = p - cell * TH;
            const unsigned short* hw = HWhs + (cell * TH + t);   // transposed: stride 384
            const float* qq = qs + cell * HID;
            float s = 0.0f;
            for (int h = 0; h < HID; ++h)
                s += tanh_fast(bf2f(hw[(size_t)h * (CELLS * TH)]) + qq[h]) * vls[h];
            scs[cell * TH + t] = s;
        }
        __syncthreads();

        // -- softmax over t (one wave per cell) --
        {
            int cell = wave;
            float s0 = scs[cell * TH + lane];
            float s1 = (lane < 16) ? scs[cell * TH + 32 + lane] : -1e30f;
            float m = fmaxf(s0, s1);
            #pragma unroll
            for (int off = 16; off >= 1; off >>= 1) m = fmaxf(m, __shfl_xor(m, off, 32));
            float e0 = __expf(s0 - m);
            float e1 = (lane < 16) ? __expf(s1 - m) : 0.0f;
            float sum = e0 + e1;
            #pragma unroll
            for (int off = 16; off >= 1; off >>= 1) sum += __shfl_xor(sum, off, 32);
            float inv = __builtin_amdgcn_rcpf(sum);
            alps[cell * TH + lane] = e0 * inv;
            if (lane < 16) alps[cell * TH + 32 + lane] = e1 * inv;
        }
        __syncthreads();

        // -- ctx = sum_t alpha * H; becomes next-step carry h (packed bf16) --
        for (int e = tid; e < CELLS * HID / 2; e += NTHR) {
            int cell = e >> 6, h = (e & 63) * 2;
            const float* al = alps + cell * TH;
            const unsigned short* hp = Hs + cell * TH * HID + h;
            float s0 = 0.0f, s1 = 0.0f;
            #pragma unroll 4
            for (int t = 0; t < TH; ++t) {
                unsigned pr = *(const unsigned*)(hp + t * HID);
                float a = al[t];
                s0 += a * bf2f((unsigned short)(pr & 0xFFFFu));
                s1 += a * bf2f((unsigned short)(pr >> 16));
            }
            ctxs[cell * HID + h]     = s0;
            ctxs[cell * HID + h + 1] = s1;
            hnf[cell * HID + h]      = s0;
            hnf[cell * HID + h + 1]  = s1;
            unsigned pk = (unsigned)f2bf(s0) | ((unsigned)f2bf(s1) << 16);
            *(unsigned*)(hnb + cell * HID + h) = pk;   // rows 8..15 remain zero
        }
        __syncthreads();

        // -- xn_new = ctx @ fc_w^T + fc_b ; emit prediction (one wave/cell) --
        {
            int cell = wave;
            float part = 0.0f;
            #pragma unroll
            for (int j = 0; j < 4; ++j) {
                int h = lane + 32 * j;
                part += ctxs[cell * HID + h] * fcw[h];
            }
            #pragma unroll
            for (int off = 16; off >= 1; off >>= 1) part += __shfl_xor(part, off, 32);
            if (lane == 0) {
                float xv = part + fcb[0];
                xns[cell] = xv;
                out[((size_t)b * TF + step) * CC + c0 + cell] = xv;
            }
        }
        __syncthreads();
    }
}

// ---------------------------------------------------------------------------
extern "C" void kernel_launch(void* const* d_in, const int* in_sizes, int n_in,
                              void* d_out, int out_size, void* d_ws, size_t ws_size,
                              hipStream_t stream) {
    const float* X   = (const float*)d_in[0];
    const float* H   = (const float*)d_in[1];
    const float* xn  = (const float*)d_in[2];
    const float* emb = (const float*)d_in[3];
    const float* Wih = (const float*)d_in[4];
    const float* Whh = (const float*)d_in[5];
    const float* bih = (const float*)d_in[6];
    const float* bhh = (const float*)d_in[7];
    const float* Wh  = (const float*)d_in[8];
    const float* Ws  = (const float*)d_in[9];
    const float* av  = (const float*)d_in[10];
    const float* fcw = (const float*)d_in[11];
    const float* fcb = (const float*)d_in[12];
    unsigned short* wsw = (unsigned short*)d_ws;
    float* out = (float*)d_out;

    prep_swizzle<<<(SW_TOTAL + NTHR - 1) / NTHR, NTHR, 0, stream>>>(Whh, Ws, Wh, wsw);

    (void)hipFuncSetAttribute((const void*)decoder_step_kernel,
                              hipFuncAttributeMaxDynamicSharedMemorySize, SMEM_BYTES);
    decoder_step_kernel<<<NWG, NTHR, SMEM_BYTES, stream>>>(
        X, H, xn, emb, Wih, bih, bhh, av, fcw, fcb, wsw, out);
}
